// GPT_Network_23751169147336
// MI455X (gfx1250) — matrix operand order repaired and checked by reading, not compiled, not run on hardware
//
#include <hip/hip_runtime.h>
#include <math.h>

typedef __attribute__((ext_vector_type(16))) _Float16 v16h;
typedef __attribute__((ext_vector_type(8)))  _Float16 v8h;
typedef __attribute__((ext_vector_type(4)))  _Float16 v4h;
typedef __attribute__((ext_vector_type(8)))  float    v8f;
typedef __attribute__((ext_vector_type(4)))  int      v4i;

#define N_LAYERS 6
#define N_HEADS  16
#define HIDDEN   1024
#define HEAD     64
#define FFWD     4096
#define VOCAB    32000
#define WIN      512
#define EMBED    128
#define BATCH    4
#define SEQ      (2 * WIN)            // 1024
#define MFULL    (BATCH * SEQ)        // 4096 rows (all positions)
#define MHALF    (BATCH * WIN)        // 2048 rows (current positions)

// CDNA5 async global->LDS copy path (ASYNCcnt), if the toolchain exposes it.
#if defined(__has_builtin)
#if __has_builtin(__builtin_amdgcn_global_load_async_to_lds_b128) && \
    __has_builtin(__builtin_amdgcn_s_wait_asynccnt)
#define USE_ASYNC_COPY 1
#endif
#endif
#ifndef USE_ASYNC_COPY
#define USE_ASYNC_COPY 0
#endif

union Frag { v16h v; v8h p[2]; };

__device__ __forceinline__ v4h f4_to_h4(float4 f) {
    v4h h;
    h[0] = (_Float16)f.x; h[1] = (_Float16)f.y;
    h[2] = (_Float16)f.z; h[3] = (_Float16)f.w;
    return h;
}

// ---------------------------------------------------------------------------
// Weight convert+transpose: f32 (K,N) -> f16 (N,K), tiled 64x64 through LDS.
// ---------------------------------------------------------------------------
__global__ __launch_bounds__(256) void wconv_kernel(const float* __restrict__ src,
                                                    _Float16* __restrict__ dst,
                                                    int K, int N, float scale)
{
    __shared__ _Float16 T[64][72];     // [k][n]
    const int tid = threadIdx.x;
    const int n0 = blockIdx.x << 6;
    const int k0 = blockIdx.y << 6;

    #pragma unroll
    for (int p = 0; p < 4; p++) {
        int r = (tid >> 4) + (p << 4);
        int g = tid & 15;
        float4 s = *(const float4*)&src[(long)(k0 + r) * N + n0 + (g << 2)];
        s.x *= scale; s.y *= scale; s.z *= scale; s.w *= scale;
        *(v4h*)&T[r][g << 2] = f4_to_h4(s);
    }
    __syncthreads();
    #pragma unroll
    for (int p = 0; p < 4; p++) {
        int n  = (tid >> 4) + (p << 4);
        int kg = tid & 15;
        v4h h;
        h[0] = T[(kg << 2) + 0][n];
        h[1] = T[(kg << 2) + 1][n];
        h[2] = T[(kg << 2) + 2][n];
        h[3] = T[(kg << 2) + 3][n];
        *(v4h*)&dst[(long)(n0 + n) * K + k0 + (kg << 2)] = h;
    }
}

// ---------------------------------------------------------------------------
__global__ __launch_bounds__(128) void embed_kernel(const int* __restrict__ ids,
                                                    const float* __restrict__ Wemb,
                                                    float* __restrict__ tok)
{
    int row = blockIdx.x;
    int e   = threadIdx.x;
    int id  = ids[row];
    tok[(long)row * EMBED + e] = Wemb[(long)id * EMBED + e] * 0.03125f; // 1024^-0.5
}

// ---------------------------------------------------------------------------
__global__ __launch_bounds__(256) void buildx_kernel(const float* __restrict__ prev,
                                                     const float* __restrict__ h,
                                                     const float* __restrict__ pos,
                                                     float* __restrict__ x)
{
    long i = (long)blockIdx.x * 256 + threadIdx.x;
    if (i >= (long)BATCH * SEQ * HIDDEN) return;
    int  c  = (int)(i % HIDDEN);
    long bt = i / HIDDEN;
    int  t  = (int)(bt % SEQ);
    int  b  = (int)(bt / SEQ);
    float v = pos[(long)t * HIDDEN + c];
    if (t < WIN) v += prev[((long)b * WIN + t) * HIDDEN + c];
    else         v += h[((long)b * WIN + (t - WIN)) * HIDDEN + c];
    x[i] = v;
}

// ---------------------------------------------------------------------------
// WMMA GEMM: C(M,N) = A(M,K) @ Bt(N,K)^T [+bias][relu]
// Exact shapes: M%128==0, N%64==0, K%32==0 (no guards -> vector staging).
// Block tile 128x64, 4 waves, each wave 32x64 (2x4 WMMA -> 1.5 LDS loads/WMMA).
// Double-buffered LDS, software-pipelined A loads; B tiles staged with CDNA5
// async global->LDS b128 copies (ASYNCcnt) when available.
// amode==1: A row m -> (m/512)*1024 + 512 + m%512 (current positions only).
// ---------------------------------------------------------------------------
__global__ __launch_bounds__(128) void gemm_wmma_kernel(const float* __restrict__ A,
                                                        const _Float16* __restrict__ Bt,
                                                        float* __restrict__ C,
                                                        const float* __restrict__ bias,
                                                        int M, int N, int K,
                                                        int amode, int relu)
{
    __shared__ _Float16 As[2][128][40];  // [buf][m][k]
    __shared__ _Float16 Bs[2][64][40];   // [buf][n][k]

    const int tid  = threadIdx.x;
    const int wave = tid >> 5;
    const int lane = tid & 31;
    const int half = lane >> 4;
    const int l15  = lane & 15;
    const int wr   = wave << 5;          // wave row offset (0/32/64/96)
    const int m0   = blockIdx.y << 7;
    const int n0   = blockIdx.x << 6;

    v8f acc[2][4];
    #pragma unroll
    for (int i = 0; i < 2; i++)
        #pragma unroll
        for (int j = 0; j < 4; j++)
            #pragma unroll
            for (int r = 0; r < 8; r++) acc[i][j][r] = 0.0f;

    float4 aReg[8];
#if !USE_ASYNC_COPY
    v8h    bReg[2];
#endif

    auto arowOf = [&](int gm) -> long {
        return (amode == 0) ? (long)gm
                            : ((long)(gm >> 9) * SEQ + WIN + (gm & 511));
    };
    auto loadA = [&](int k0) {
        #pragma unroll
        for (int p = 0; p < 8; p++) {
            int r = (tid >> 3) + (p << 4);
            int g = tid & 7;
            aReg[p] = *(const float4*)&A[arowOf(m0 + r) * (long)K + k0 + (g << 2)];
        }
    };
    auto storeA = [&](int buf) {
        #pragma unroll
        for (int p = 0; p < 8; p++) {
            int r = (tid >> 3) + (p << 4);
            int g = tid & 7;
            *(v4h*)&As[buf][r][g << 2] = f4_to_h4(aReg[p]);
        }
    };
#if USE_ASYNC_COPY
    auto stageB = [&](int k0, int buf) {   // async global->LDS, no VGPR round trip
        #pragma unroll
        for (int p = 0; p < 2; p++) {
            int r = (tid >> 2) + (p << 5);
            int g = tid & 3;
            __builtin_amdgcn_global_load_async_to_lds_b128(
                (v4i*)(void*)&Bt[(long)(n0 + r) * K + k0 + (g << 3)],
                (v4i*)(void*)&Bs[buf][r][g << 3],
                0, 0);
        }
    };
#else
    auto loadB = [&](int k0) {
        #pragma unroll
        for (int p = 0; p < 2; p++) {
            int r = (tid >> 2) + (p << 5);
            int g = tid & 3;
            bReg[p] = *(const v8h*)&Bt[(long)(n0 + r) * K + k0 + (g << 3)];
        }
    };
    auto storeB = [&](int buf) {
        #pragma unroll
        for (int p = 0; p < 2; p++) {
            int r = (tid >> 2) + (p << 5);
            int g = tid & 3;
            *(v8h*)&Bs[buf][r][g << 3] = bReg[p];
        }
    };
#endif

    // prologue: fill buffer 0
    loadA(0);
#if USE_ASYNC_COPY
    stageB(0, 0);
    storeA(0);
    __builtin_amdgcn_s_wait_asynccnt(0);
#else
    loadB(0);
    storeA(0);
    storeB(0);
#endif
    __syncthreads();

    for (int k0 = 0; k0 < K; k0 += 32) {
        const int  buf  = (k0 >> 5) & 1;
        const bool more = (k0 + 32) < K;
        if (more) {
            loadA(k0 + 32);                 // overlap with WMMAs below
#if USE_ASYNC_COPY
            stageB(k0 + 32, buf ^ 1);       // async copy straight into other buffer
#else
            loadB(k0 + 32);
#endif
        }
        if (k0 + 64 < K) {                  // L2 prefetch of tile after next
            __builtin_prefetch(&A[arowOf(m0 + tid) * (long)K + k0 + 64], 0, 3);
            if (tid < 64)
                __builtin_prefetch(&Bt[(long)(n0 + tid) * K + k0 + 64], 0, 3);
        }

        Frag af[2], bf[4];
        #pragma unroll
        for (int i = 0; i < 2; i++) {
            const int am = wr + (i << 4) + l15;
            af[i].p[0] = *(const v8h*)&As[buf][am][half << 3];
            af[i].p[1] = *(const v8h*)&As[buf][am][16 + (half << 3)];
        }
        #pragma unroll
        for (int j = 0; j < 4; j++) {
            const int bn = (j << 4) + l15;
            bf[j].p[0] = *(const v8h*)&Bs[buf][bn][half << 3];
            bf[j].p[1] = *(const v8h*)&Bs[buf][bn][16 + (half << 3)];
        }
        #pragma unroll
        for (int i = 0; i < 2; i++)
            #pragma unroll
            for (int j = 0; j < 4; j++)
                acc[i][j] = __builtin_amdgcn_wmma_f32_16x16x32_f16(false, af[i].v, false, bf[j].v,
                                                                   (short)0, acc[i][j], false, false);

        if (more) {
            storeA(buf ^ 1);
#if USE_ASYNC_COPY
            __builtin_amdgcn_s_wait_asynccnt(0);
#else
            storeB(buf ^ 1);
#endif
        }
        __syncthreads();
    }

    #pragma unroll
    for (int j = 0; j < 4; j++) {
        const int gn = n0 + (j << 4) + l15;
        const float bv = bias ? bias[gn] : 0.0f;
        #pragma unroll
        for (int i = 0; i < 2; i++) {
            #pragma unroll
            for (int r = 0; r < 8; r++) {
                const int gm = m0 + wr + (i << 4) + (half << 3) + r;
                float v = acc[i][j][r] + bv;
                if (relu) v = fmaxf(v, 0.0f);
                C[(long)gm * N + gn] = v;
            }
        }
    }
}

// ---------------------------------------------------------------------------
// Flash attention (causal, q rows 512..1023 only).
// grid = (8 q-tiles, 16 heads, 4 batch), block = 128 (4 waves).
// ---------------------------------------------------------------------------
__global__ __launch_bounds__(128) void flash_kernel(const float* __restrict__ q,
                                                    const float* __restrict__ k,
                                                    const float* __restrict__ v,
                                                    float* __restrict__ o)
{
    __shared__ _Float16 Qs[64][72];
    __shared__ _Float16 Ks[64][72];
    __shared__ _Float16 Vt[64][72];    // [d][key] transposed
    __shared__ _Float16 Ps[64][72];
    __shared__ float    Ss[64][65];
    __shared__ float    rowM[64], rowL[64], rowA[64];
    __shared__ float    pred[64][2];

    const int tid  = threadIdx.x;
    const int wave = tid >> 5;
    const int lane = tid & 31;
    const int half = lane >> 4;
    const int l15  = lane & 15;
    const int qt   = blockIdx.x;
    const int hh   = blockIdx.y;
    const int b    = blockIdx.z;
    const int qg0  = (8 + qt) * 64;

    #pragma unroll
    for (int p = 0; p < 8; p++) {
        int r = (tid >> 4) + (p << 3);
        int g = tid & 15;
        float4 qv = *(const float4*)&q[(((long)b * WIN + qt * 64 + r) * HIDDEN) + hh * HEAD + (g << 2)];
        *(v4h*)&Qs[r][g << 2] = f4_to_h4(qv);
    }
    if (tid < 64) { rowM[tid] = -1.0e30f; rowL[tid] = 0.0f; }

    v8f oacc[4];
    #pragma unroll
    for (int j = 0; j < 4; j++)
        #pragma unroll
        for (int r = 0; r < 8; r++) oacc[j][r] = 0.0f;

    const int nkt = 9 + qt;
    for (int kt = 0; kt < nkt; kt++) {
        __syncthreads();
        #pragma unroll
        for (int p = 0; p < 8; p++) {
            int r = (tid >> 4) + (p << 3);
            int g = tid & 15;
            long base = (((long)b * SEQ + kt * 64 + r) * HIDDEN) + hh * HEAD + (g << 2);
            float4 kv = *(const float4*)&k[base];
            float4 vv = *(const float4*)&v[base];
            *(v4h*)&Ks[r][g << 2] = f4_to_h4(kv);
            Vt[(g << 2) + 0][r] = (_Float16)vv.x;
            Vt[(g << 2) + 1][r] = (_Float16)vv.y;
            Vt[(g << 2) + 2][r] = (_Float16)vv.z;
            Vt[(g << 2) + 3][r] = (_Float16)vv.w;
        }
        __syncthreads();

        // S = Q x K^T : 16 q rows x 64 keys per wave
        v8f sacc[4];
        #pragma unroll
        for (int j = 0; j < 4; j++)
            #pragma unroll
            for (int r = 0; r < 8; r++) sacc[j][r] = 0.0f;

        #pragma unroll
        for (int kk = 0; kk < 64; kk += 32) {
            Frag af;
            const int qm = (wave << 4) + l15;
            af.p[0] = *(const v8h*)&Qs[qm][kk + (half << 3)];
            af.p[1] = *(const v8h*)&Qs[qm][kk + 16 + (half << 3)];
            #pragma unroll
            for (int j = 0; j < 4; j++) {
                Frag bf;
                const int kn = (j << 4) + l15;
                bf.p[0] = *(const v8h*)&Ks[kn][kk + (half << 3)];
                bf.p[1] = *(const v8h*)&Ks[kn][kk + 16 + (half << 3)];
                sacc[j] = __builtin_amdgcn_wmma_f32_16x16x32_f16(false, af.v, false, bf.v,
                                                                 (short)0, sacc[j], false, false);
            }
        }

        #pragma unroll
        for (int j = 0; j < 4; j++) {
            int kl = (j << 4) + l15;
            int kg = kt * 64 + kl;
            #pragma unroll
            for (int r = 0; r < 8; r++) {
                int ql = (wave << 4) + (half << 3) + r;
                float s = sacc[j][r];
                if (kg > qg0 + ql) s = -1.0e9f;
                Ss[ql][kl] = s;
            }
        }
        __syncthreads();

        // online softmax: 2 threads per q row (32 cols each)
        const int row = tid >> 1;
        const int hx  = tid & 1;
        const int c0  = hx << 5;
        {
            float mloc = -1.0e30f;
            #pragma unroll
            for (int c = 0; c < 32; c++) mloc = fmaxf(mloc, Ss[row][c0 + c]);
            pred[row][hx] = mloc;
        }
        __syncthreads();
        const float mOld = rowM[row];
        const float mNew = fmaxf(mOld, fmaxf(pred[row][0], pred[row][1]));
        float ls = 0.0f;
        #pragma unroll
        for (int c = 0; c < 32; c++) {
            float pv = __expf(Ss[row][c0 + c] - mNew);
            Ps[row][c0 + c] = (_Float16)pv;
            ls += pv;
        }
        __syncthreads();
        pred[row][hx] = ls;
        __syncthreads();
        if (hx == 0) {
            float alpha = __expf(mOld - mNew);
            rowA[row] = alpha;
            rowL[row] = rowL[row] * alpha + pred[row][0] + pred[row][1];
            rowM[row] = mNew;
        }
        __syncthreads();

        // O = O*alpha + P @ V
        #pragma unroll
        for (int j = 0; j < 4; j++)
            #pragma unroll
            for (int r = 0; r < 8; r++)
                oacc[j][r] *= rowA[(wave << 4) + (half << 3) + r];

        #pragma unroll
        for (int kk = 0; kk < 64; kk += 32) {
            Frag af;
            const int qm = (wave << 4) + l15;
            af.p[0] = *(const v8h*)&Ps[qm][kk + (half << 3)];
            af.p[1] = *(const v8h*)&Ps[qm][kk + 16 + (half << 3)];
            #pragma unroll
            for (int j = 0; j < 4; j++) {
                Frag bf;
                const int dn = (j << 4) + l15;
                bf.p[0] = *(const v8h*)&Vt[dn][kk + (half << 3)];
                bf.p[1] = *(const v8h*)&Vt[dn][kk + 16 + (half << 3)];
                oacc[j] = __builtin_amdgcn_wmma_f32_16x16x32_f16(false, af.v, false, bf.v,
                                                                 (short)0, oacc[j], false, false);
            }
        }
    }
    __syncthreads();

    #pragma unroll
    for (int j = 0; j < 4; j++) {
        int d = (j << 4) + l15;
        #pragma unroll
        for (int r = 0; r < 8; r++) {
            int ql = (wave << 4) + (half << 3) + r;
            float val = oacc[j][r] / rowL[ql];
            o[(((long)b * WIN + qt * 64 + ql) * HIDDEN) + hh * HEAD + d] = val;
        }
    }
}

// ---------------------------------------------------------------------------
__global__ __launch_bounds__(256) void addln_kernel(const float* __restrict__ inp,
                                                    const float* __restrict__ res, int res_mode,
                                                    const float* __restrict__ bias,
                                                    const float* __restrict__ scale,
                                                    float* __restrict__ out,
                                                    float* __restrict__ out2)
{
    __shared__ float red[256];
    const int row = blockIdx.x;
    const int tid = threadIdx.x;
    const float* ip = inp + (long)row * HIDDEN;
    long rr = res_mode ? ((long)(row >> 9) * SEQ + WIN + (row & 511)) : (long)row;
    const float* rp = res + rr * HIDDEN;

    float s = 0.0f;
    for (int c = tid; c < HIDDEN; c += 256) s += ip[c];
    red[tid] = s; __syncthreads();
    for (int st = 128; st > 0; st >>= 1) { if (tid < st) red[tid] += red[tid + st]; __syncthreads(); }
    const float mean = red[0] * (1.0f / HIDDEN);
    __syncthreads();

    float vv = 0.0f;
    for (int c = tid; c < HIDDEN; c += 256) { float d = ip[c] - mean; vv += d * d; }
    red[tid] = vv; __syncthreads();
    for (int st = 128; st > 0; st >>= 1) { if (tid < st) red[tid] += red[tid + st]; __syncthreads(); }
    const float rstd = rsqrtf(red[0] * (1.0f / HIDDEN) + 1e-6f);

    for (int c = tid; c < HIDDEN; c += 256) {
        float val = rp[c] + (ip[c] - mean) * rstd * scale[c] + bias[c];
        out[(long)row * HIDDEN + c] = val;
        if (out2) out2[(long)row * HIDDEN + c] = val;
    }
}

// ---------------------------------------------------------------------------
extern "C" void kernel_launch(void* const* d_in, const int* in_sizes, int n_in,
                              void* d_out, int out_size, void* d_ws, size_t ws_size,
                              hipStream_t stream)
{
    (void)in_sizes; (void)n_in; (void)out_size; (void)ws_size;

    const int*   x_input = (const int*)  d_in[0];
    const float* prev_in = (const float*)d_in[1];
    const float* W_emb   = (const float*)d_in[2];
    const float* W_lin   = (const float*)d_in[3];
    const float* p_dec   = (const float*)d_in[4];
    const float* p_q     = (const float*)d_in[5];
    const float* p_k     = (const float*)d_in[6];
    const float* p_v     = (const float*)d_in[7];
    const float* p_c     = (const float*)d_in[8];
    const float* p_f1    = (const float*)d_in[9];
    const float* p_f2    = (const float*)d_in[10];
    const float* b_f1    = (const float*)d_in[11];
    const float* b_f2    = (const float*)d_in[12];
    const float* b_b1    = (const float*)d_in[13];
    const float* b_b2    = (const float*)d_in[14];
    const float* b_s1    = (const float*)d_in[15];
    const float* b_s2    = (const float*)d_in[16];
    const float* pos_all = (const float*)d_in[17];

    char* wsp = (char*)d_ws;
    auto alloc = [&](size_t bytes) -> char* {
        char* p = wsp;
        wsp += (bytes + 255) & ~(size_t)255;
        return p;
    };

    _Float16* wlin16 = (_Float16*)alloc((size_t)HIDDEN * EMBED * 2);
    _Float16* wq16   = (_Float16*)alloc((size_t)N_LAYERS * HIDDEN * HIDDEN * 2);
    _Float16* wk16   = (_Float16*)alloc((size_t)N_LAYERS * HIDDEN * HIDDEN * 2);
    _Float16* wv16   = (_Float16*)alloc((size_t)N_LAYERS * HIDDEN * HIDDEN * 2);
    _Float16* wc16   = (_Float16*)alloc((size_t)N_LAYERS * HIDDEN * HIDDEN * 2);
    _Float16* wf116  = (_Float16*)alloc((size_t)N_LAYERS * HIDDEN * FFWD * 2);
    _Float16* wf216  = (_Float16*)alloc((size_t)N_LAYERS * FFWD * HIDDEN * 2);
    _Float16* wdec16 = (_Float16*)alloc((size_t)VOCAB * HIDDEN * 2);

    float* tok   = (float*)alloc((size_t)MHALF * EMBED * 4);
    float* hbuf  = (float*)alloc((size_t)MHALF * HIDDEN * 4);
    float* xbuf  = (float*)alloc((size_t)MFULL * HIDDEN * 4);
    float* qb    = (float*)alloc((size_t)MHALF * HIDDEN * 4);
    float* kb    = (float*)alloc((size_t)MFULL * HIDDEN * 4);
    float* vb    = (float*)alloc((size_t)MFULL * HIDDEN * 4);
    float* ob    = (float*)alloc((size_t)MHALF * HIDDEN * 4);
    float* attnb = (float*)alloc((size_t)MHALF * HIDDEN * 4);
    float* xself = (float*)alloc((size_t)MHALF * HIDDEN * 4);
    float* ff1b  = (float*)alloc((size_t)MHALF * FFWD * 4);
    float* ffb   = (float*)alloc((size_t)MHALF * HIDDEN * 4);

    auto conv = [&](const float* src, _Float16* dst, int K, int N, float sc) {
        dim3 g(N / 64, K / 64);
        wconv_kernel<<<g, 256, 0, stream>>>(src, dst, K, N, sc);
    };
    auto gemm = [&](const float* A, const _Float16* Bt, float* C, const float* bias,
                    int M, int N, int K, int amode, int relu) {
        dim3 g(N / 64, M / 128);
        gemm_wmma_kernel<<<g, 128, 0, stream>>>(A, Bt, C, bias, M, N, K, amode, relu);
    };

    conv(W_lin, wlin16, EMBED, HIDDEN, 1.0f);
    conv(p_dec, wdec16, HIDDEN, VOCAB, 1.0f);
    for (int l = 0; l < N_LAYERS; l++) {
        size_t w1 = (size_t)l * HIDDEN * HIDDEN;
        conv(p_q  + w1, wq16  + w1, HIDDEN, HIDDEN, 0.125f);  // fold HEAD^-0.5
        conv(p_k  + w1, wk16  + w1, HIDDEN, HIDDEN, 1.0f);
        conv(p_v  + w1, wv16  + w1, HIDDEN, HIDDEN, 1.0f);
        conv(p_c  + w1, wc16  + w1, HIDDEN, HIDDEN, 1.0f);
        size_t w2 = (size_t)l * HIDDEN * FFWD;
        conv(p_f1 + w2, wf116 + w2, HIDDEN, FFWD, 1.0f);
        conv(p_f2 + w2, wf216 + w2, FFWD, HIDDEN, 1.0f);
    }

    embed_kernel<<<MHALF, 128, 0, stream>>>(x_input, W_emb, tok);
    gemm(tok, wlin16, hbuf, nullptr, MHALF, HIDDEN, EMBED, 0, 0);

    float* logits  = (float*)d_out;
    float* dec_out = (float*)d_out + (size_t)BATCH * WIN * VOCAB;

    for (int l = 0; l < N_LAYERS; l++) {
        size_t w1 = (size_t)l * HIDDEN * HIDDEN;
        size_t w2 = (size_t)l * HIDDEN * FFWD;

        int bx = (int)(((long)BATCH * SEQ * HIDDEN + 255) / 256);
        buildx_kernel<<<bx, 256, 0, stream>>>(prev_in + (size_t)l * BATCH * WIN * HIDDEN,
                                              hbuf, pos_all + (size_t)l * SEQ * HIDDEN, xbuf);

        gemm(xbuf, wq16 + w1, qb, nullptr, MHALF, HIDDEN, HIDDEN, 1, 0);
        gemm(xbuf, wk16 + w1, kb, nullptr, MFULL, HIDDEN, HIDDEN, 0, 0);
        gemm(xbuf, wv16 + w1, vb, nullptr, MFULL, HIDDEN, HIDDEN, 0, 0);

        flash_kernel<<<dim3(8, N_HEADS, BATCH), 128, 0, stream>>>(qb, kb, vb, ob);

        gemm(ob, wc16 + w1, attnb, nullptr, MHALF, HIDDEN, HIDDEN, 0, 0);
        addln_kernel<<<MHALF, 256, 0, stream>>>(attnb, xbuf, 1,
                                                b_b1 + (size_t)l * HIDDEN,
                                                b_s1 + (size_t)l * HIDDEN, xself, nullptr);

        gemm(xself, wf116 + w2, ff1b, b_f1 + (size_t)l * FFWD, MHALF, FFWD, HIDDEN, 0, 1);
        gemm(ff1b,  wf216 + w2, ffb,  b_f2 + (size_t)l * HIDDEN, MHALF, HIDDEN, FFWD, 0, 0);

        addln_kernel<<<MHALF, 256, 0, stream>>>(ffb, xself, 0,
                                                b_b2 + (size_t)l * HIDDEN,
                                                b_s2 + (size_t)l * HIDDEN, hbuf,
                                                dec_out + (size_t)l * MHALF * HIDDEN);
    }

    gemm(hbuf, wdec16, logits, nullptr, MHALF, VOCAB, HIDDEN, 0, 0);
}
